// FlowNetMotionEncoder_14259291423007
// MI455X (gfx1250) — compile-verified, matrix operand order checked by reading, not run on hardware
//
#include <hip/hip_runtime.h>
#include <hip/hip_bf16.h>

typedef __attribute__((ext_vector_type(16))) _Float16 v16h;
typedef __attribute__((ext_vector_type(8)))  float    v8f;

#define BN_EPS 1e-5f

// ---------------- CDNA5 async-to-LDS support (guarded) ----------------
#if __has_builtin(__builtin_amdgcn_global_load_async_to_lds_b128)
#define HAVE_ASYNC 1
#else
#define HAVE_ASYNC 0
#endif

typedef int v4i_ __attribute__((__vector_size__(16)));
typedef __attribute__((address_space(1))) v4i_ gv4;   // global int4
typedef __attribute__((address_space(3))) v4i_ lv4;   // LDS int4

__device__ __forceinline__ gv4* to_gbl4(const void* p) {
  return (gv4*)(unsigned long long)p;
}
__device__ __forceinline__ lv4* to_lds4(void* p) {
  // generic pointers to LDS carry the LDS byte offset in their low 32 bits
  return (lv4*)(unsigned int)(unsigned long long)p;
}

__device__ __forceinline__ void wait_async0() {
#if HAVE_ASYNC
#if __has_builtin(__builtin_amdgcn_s_wait_asynccnt)
  __builtin_amdgcn_s_wait_asynccnt(0);
#else
  asm volatile("s_wait_asynccnt 0" ::: "memory");
#endif
#endif
}

// ------------------------------------------------------------------
// Fold BN into conv weights, convert to f16, pad K to multiple of 32.
// wp layout: [Cout][Kpad] row-major (K = ci*kh*kw + r*kw + s, OIHW order).
// ------------------------------------------------------------------
__global__ void prep_weights(const float* __restrict__ w,
                             const float* __restrict__ b,
                             const float* __restrict__ gamma,
                             const float* __restrict__ beta,
                             const float* __restrict__ mean,
                             const float* __restrict__ var,
                             _Float16* __restrict__ wp,
                             float* __restrict__ bias2,
                             int Cout, int K, int Kpad) {
  int idx = blockIdx.x * blockDim.x + threadIdx.x;
  if (idx >= Cout * Kpad) return;
  int co = idx / Kpad;
  int k  = idx - co * Kpad;
  float scale = gamma[co] * rsqrtf(var[co] + BN_EPS);
  _Float16 v = (_Float16)0.f;
  if (k < K) v = (_Float16)(w[(size_t)co * K + k] * scale);
  wp[idx] = v;
  if (k == 0) bias2[co] = (b[co] - mean[co]) * scale + beta[co];
}

__global__ void cvt_f32_f16(const float* __restrict__ in,
                            _Float16* __restrict__ out, int n) {
  int i = blockIdx.x * blockDim.x + threadIdx.x;
  if (i < n) out[i] = (_Float16)in[i];
}

// ------------------------------------------------------------------
// Implicit-GEMM conv + bias + ReLU using v_wmma_f32_16x16x32_f16.
//   GEMM:  M = Cout (A = folded weights), N = output pixels (B = im2col),
//          K = Cin*KH*KW, Kpad = K rounded up to 32.
//   Block = 128 threads = 4 wave32; each wave computes a 32x32 C subtile.
//   Double-buffered LDS; weights staged via GLOBAL_LOAD_ASYNC_TO_LDS_B128
//   (ASYNCcnt) so next-chunk copies overlap current-chunk WMMAs.
// ------------------------------------------------------------------
#define TBM 64
#define TBN 64
#define TBK 32
#define KP  40   // padded LDS row stride (halves); 80B rows, 16B chunks
#define TILE_HALVES (TBM * KP)

union Frag { v16h h; uint4 q[2]; };

__global__ __launch_bounds__(128) void conv_wmma(
    const _Float16* __restrict__ src,   // [Nimg(+grp skip)][Cin][H][W]
    const _Float16* __restrict__ wgt,   // [Cout][Kpad] f16
    const float*    __restrict__ bias,  // [Cout]
    _Float16*       __restrict__ dst,   // [Nimg][CtotDst][OH][OW]
    int Nimg, int Cin, int H, int W,
    int Cout, int KH, int KW, int strd, int pad,
    int OH, int OW, int K, int Kpad,
    int CtotDst, int Coff, int srcGroup, int doRelu)
{
  __shared__ __align__(16) _Float16 WtS[2 * TILE_HALVES];
  __shared__ __align__(16) _Float16 ImS[2 * TILE_HALVES];

  const int tid   = threadIdx.x;
  const int lane  = tid & 31;
  const int wave  = tid >> 5;
  const int wm    = wave >> 1;      // cout direction within block (0..1)
  const int wn    = wave & 1;       // pixel direction within block (0..1)
  const int lHalf = lane & 15;
  const int hi    = lane >> 4;      // K half-group per WMMA 16-bit layout

  const int pixBase = blockIdx.x * TBN;
  const int coBase  = blockIdx.y * TBM;
  const int Mtotal  = Nimg * OH * OW;
  const int KHW     = KH * KW;

  // staging coordinates: each thread fills 16 halves of one LDS row
  const int sRow = tid >> 1;           // 0..63
  const int sKo  = (tid & 1) * 16;     // 0 or 16

  // pixel decomposition for the im2col row this thread stages
  int m = pixBase + sRow;
  bool mValid = m < Mtotal;
  int oy = 0, ox = 0, iBase = 0;
  if (mValid) {
    int n_img = m / (OH * OW);
    int rem   = m - n_img * (OH * OW);
    oy = rem / OW;
    ox = rem - oy * OW;
    int actual = n_img + (srcGroup ? (n_img / srcGroup) : 0);
    iBase = actual * Cin;
  }
  const int  wRow   = coBase + sRow;
  const bool wValid = wRow < Cout;

  // ---- weight tile staging (async DMA to LDS when available) ----
  auto stageA = [&](int kk, _Float16* buf) {
    _Float16* l = buf + sRow * KP + sKo;
    if (wValid) {
      const _Float16* g = wgt + (size_t)wRow * Kpad + kk + sKo;
#if HAVE_ASYNC
      __builtin_amdgcn_global_load_async_to_lds_b128(to_gbl4(g), to_lds4(l), 0, 0);
      __builtin_amdgcn_global_load_async_to_lds_b128(to_gbl4(g + 8), to_lds4(l + 8), 0, 0);
#else
      uint4 q0 = ((const uint4*)g)[0];
      uint4 q1 = ((const uint4*)g)[1];
      ((uint4*)l)[0] = q0;
      ((uint4*)l)[1] = q1;
#endif
    } else {
      uint4 z = {0u, 0u, 0u, 0u};
      ((uint4*)l)[0] = z;
      ((uint4*)l)[1] = z;
    }
  };

  // ---- im2col tile staging: gather 16 halves, incremental (ci,r,s) ----
  auto stageB = [&](int kk, _Float16* buf) {
    __align__(16) _Float16 vals[16];
    int kIdx = kk + sKo;
    int ci = kIdx / KHW;
    int rs = kIdx - ci * KHW;
    int r  = rs / KW;
    int s  = rs - r * KW;
#pragma unroll
    for (int u = 0; u < 16; ++u) {
      _Float16 v = (_Float16)0.f;
      if (mValid && (kIdx + u) < K) {
        int iy = oy * strd - pad + r;
        int ix = ox * strd - pad + s;
        if ((unsigned)iy < (unsigned)H && (unsigned)ix < (unsigned)W)
          v = src[((size_t)(iBase + ci) * H + iy) * W + ix];
      }
      vals[u] = v;
      if (++s == KW) { s = 0; if (++r == KH) { r = 0; ++ci; } }
    }
    uint4* l = (uint4*)(buf + sRow * KP + sKo);
    l[0] = *(const uint4*)&vals[0];
    l[1] = *(const uint4*)&vals[8];
  };

  v8f acc[2][2] = {};

  stageA(0, WtS);
  stageB(0, ImS);

  for (int kk = 0; kk < Kpad; kk += TBK) {
    const int cur = (kk >> 5) & 1;
    _Float16* Wc = WtS + cur * TILE_HALVES;
    _Float16* Ic = ImS + cur * TILE_HALVES;

    wait_async0();        // this wave's async LDS writes resolved
    __syncthreads();      // whole tile visible to all waves

    int nk = kk + TBK;
    if (nk < Kpad) {      // overlap next-chunk staging with WMMAs
      stageA(nk, WtS + (cur ^ 1) * TILE_HALVES);
      stageB(nk, ImS + (cur ^ 1) * TILE_HALVES);
    }

    // ---- load fragments per CDNA5 16-bit A/B lane layout ----
    Frag a0, a1, b0, b1;
    {
      const _Float16* p = Wc + (wm * 32 + lHalf) * KP + hi * 8;
      a0.q[0] = *(const uint4*)(p);       a0.q[1] = *(const uint4*)(p + 16);
      p = Wc + (wm * 32 + 16 + lHalf) * KP + hi * 8;
      a1.q[0] = *(const uint4*)(p);       a1.q[1] = *(const uint4*)(p + 16);
      p = Ic + (wn * 32 + lHalf) * KP + hi * 8;
      b0.q[0] = *(const uint4*)(p);       b0.q[1] = *(const uint4*)(p + 16);
      p = Ic + (wn * 32 + 16 + lHalf) * KP + hi * 8;
      b1.q[0] = *(const uint4*)(p);       b1.q[1] = *(const uint4*)(p + 16);
    }
    acc[0][0] = __builtin_amdgcn_wmma_f32_16x16x32_f16(false, a0.h, false, b0.h,
                    (short)0, acc[0][0], false, false);
    acc[0][1] = __builtin_amdgcn_wmma_f32_16x16x32_f16(false, a0.h, false, b1.h,
                    (short)0, acc[0][1], false, false);
    acc[1][0] = __builtin_amdgcn_wmma_f32_16x16x32_f16(false, a1.h, false, b0.h,
                    (short)0, acc[1][0], false, false);
    acc[1][1] = __builtin_amdgcn_wmma_f32_16x16x32_f16(false, a1.h, false, b1.h,
                    (short)0, acc[1][1], false, false);
  }

  // ---- epilogue: bias + ReLU + f16 store (C/D layout: VGPR j -> M row) ----
#pragma unroll
  for (int ni = 0; ni < 2; ++ni) {
    int px = pixBase + wn * 32 + ni * 16 + lHalf;
    if (px >= Mtotal) continue;
    int pimg = px / (OH * OW);
    int prem = px - pimg * (OH * OW);
    int poy  = prem / OW;
    int pox  = prem - poy * OW;
    size_t outBase = ((size_t)pimg * CtotDst + Coff) * OH * OW
                     + (size_t)poy * OW + pox;
#pragma unroll
    for (int mi = 0; mi < 2; ++mi) {
#pragma unroll
      for (int j = 0; j < 8; ++j) {
        int co = coBase + wm * 32 + mi * 16 + hi * 8 + j;
        if (co < Cout) {
          float v = acc[mi][ni][j] + bias[co];
          if (doRelu) v = fmaxf(v, 0.f);
          dst[outBase + (size_t)co * OH * OW] = (_Float16)v;
        }
      }
    }
  }
}

// ------------------------------------------------------------------
// Cost-volume correlation: out[p, di*21+dj, y, x] =
//   (1/256) * sum_c f1[c,y,x] * f2[c, y+di-10, x+dj-10]  (zero pad)
// Block = (32 x-lanes) x (8 dj-groups); per-(p,y) f1 row tile + per-di
// padded f2 slab staged in LDS.
// ------------------------------------------------------------------
__global__ __launch_bounds__(256) void correlation_k(
    const _Float16* __restrict__ feats,  // [4][5][256][32][32]
    _Float16* __restrict__ cat)          // [16][473][32][32], ch 0..440
{
  __shared__ __align__(16) _Float16 f1s[256 * 32];
  __shared__ __align__(16) _Float16 f2s[256 * 56];
  const int p  = blockIdx.x;        // pair 0..15
  const int y  = blockIdx.y;        // 0..31
  const int x  = threadIdx.x & 31;
  const int dg = threadIdx.x >> 5;  // 0..7
  const int b  = p >> 2;
  const int t  = p & 3;
  const _Float16* f1 = feats + (size_t)(b * 5 + t) * 256 * 1024;
  const _Float16* f2 = f1 + 256 * 1024;

  { // stage f1[c, y, :] for all 256 channels (one row per thread)
    int c = threadIdx.x;
    const _Float16* g = f1 + (size_t)c * 1024 + y * 32;
    _Float16* l = f1s + c * 32;
#if HAVE_ASYNC
    __builtin_amdgcn_global_load_async_to_lds_b128(to_gbl4(g),      to_lds4(l),      0, 0);
    __builtin_amdgcn_global_load_async_to_lds_b128(to_gbl4(g + 8),  to_lds4(l + 8),  0, 0);
    __builtin_amdgcn_global_load_async_to_lds_b128(to_gbl4(g + 16), to_lds4(l + 16), 0, 0);
    __builtin_amdgcn_global_load_async_to_lds_b128(to_gbl4(g + 24), to_lds4(l + 24), 0, 0);
    wait_async0();
#else
    const uint4* gq = (const uint4*)g;
    uint4* lq = (uint4*)l;
    lq[0] = gq[0]; lq[1] = gq[1]; lq[2] = gq[2]; lq[3] = gq[3];
#endif
  }

  for (int di = 0; di < 21; ++di) {
    __syncthreads();
    { // stage f2[c, y+di-10, -10..41] with zero padding
      int c  = threadIdx.x;
      int iy = y + di - 10;
      _Float16* row = f2s + c * 56;
      if ((unsigned)iy < 32u) {
        const _Float16* grow = f2 + (size_t)c * 1024 + iy * 32;
        for (int xp = 0; xp < 52; ++xp) {
          int xi = xp - 10;
          row[xp] = ((unsigned)xi < 32u) ? grow[xi] : (_Float16)0.f;
        }
      } else {
        for (int xp = 0; xp < 52; ++xp) row[xp] = (_Float16)0.f;
      }
    }
    __syncthreads();
    for (int dj = dg; dj < 21; dj += 8) {
      float acc = 0.f;
      for (int c = 0; c < 256; ++c)
        acc += (float)f1s[c * 32 + x] * (float)f2s[c * 56 + x + dj];
      int d = di * 21 + dj;
      cat[(((size_t)p * 473 + d) * 32 + y) * 32 + x] =
          (_Float16)(acc * (1.0f / 256.0f));
    }
  }
}

// ------------------------------------------------------------------
__global__ void avgpool_k(const _Float16* __restrict__ in,
                          float* __restrict__ out, int NC, int S) {
  int i = blockIdx.x * blockDim.x + threadIdx.x;
  if (i >= NC) return;
  float s = 0.f;
  for (int k = 0; k < S; ++k) s += (float)in[(size_t)i * S + k];
  out[i] = s / (float)S;
}

__global__ void fc_k(const float* __restrict__ x, const float* __restrict__ w,
                     const float* __restrict__ b, float* __restrict__ out,
                     int M, int N, int Kd, int doRelu) {
  int i = blockIdx.x * blockDim.x + threadIdx.x;
  if (i >= M * N) return;
  int mi = i / N, n = i - mi * N;
  float acc = b[n];
  const float* xr = x + (size_t)mi * Kd;
  const float* wr = w + (size_t)n * Kd;
  for (int k = 0; k < Kd; ++k) acc += xr[k] * wr[k];
  if (doRelu) acc = fmaxf(acc, 0.f);
  out[i] = acc;
}

// ------------------------------------------------------------------
extern "C" void kernel_launch(void* const* d_in, const int* in_sizes, int n_in,
                              void* d_out, int out_size, void* d_ws,
                              size_t ws_size, hipStream_t stream) {
  (void)in_sizes; (void)n_in; (void)out_size; (void)ws_size;
  const float* images = (const float*)d_in[0];

  // layers: conv1, conv2, conv3, conv_redir, conv3_1, conv4, conv4_1,
  //         conv5, conv5_1, conv6  (d_in base = 1 + i*6: w,b,gamma,beta,mean,var)
  static const int LCout[10] = {64, 128, 256, 32, 256, 512, 512, 512, 512, 1024};
  static const int LCin [10] = {3, 64, 128, 256, 473, 256, 512, 512, 512, 512};
  static const int LKs  [10] = {7, 5, 5, 1, 3, 3, 3, 3, 3, 3};

  char* ws = (char*)d_ws;
  size_t off = 0;
  auto alloc = [&](size_t bytes) -> void* {
    size_t o = (off + 255) & ~(size_t)255;
    off = o + bytes;
    return (void*)(ws + o);
  };

  _Float16* wp[10];
  float*    bs[10];
  int Kdim[10], Kpad[10];
  for (int i = 0; i < 10; ++i) {
    Kdim[i] = LCin[i] * LKs[i] * LKs[i];
    Kpad[i] = (Kdim[i] + 31) & ~31;
    wp[i] = (_Float16*)alloc((size_t)LCout[i] * Kpad[i] * sizeof(_Float16));
    bs[i] = (float*)alloc((size_t)LCout[i] * sizeof(float));
  }
  _Float16* imgH = (_Float16*)alloc((size_t)20 * 3 * 256 * 256 * 2);
  _Float16* a1   = (_Float16*)alloc((size_t)20 * 64 * 128 * 128 * 2);
  _Float16* a2   = (_Float16*)alloc((size_t)20 * 128 * 64 * 64 * 2);
  _Float16* a3   = (_Float16*)alloc((size_t)20 * 256 * 32 * 32 * 2);
  _Float16* cat  = (_Float16*)alloc((size_t)16 * 473 * 32 * 32 * 2);
  float* pooled  = (float*)alloc((size_t)16 * 1024 * 4);
  float* fc1o    = (float*)alloc((size_t)16 * 512 * 4);

  for (int i = 0; i < 10; ++i) {
    const float* w  = (const float*)d_in[1 + i * 6 + 0];
    const float* b  = (const float*)d_in[1 + i * 6 + 1];
    const float* g  = (const float*)d_in[1 + i * 6 + 2];
    const float* be = (const float*)d_in[1 + i * 6 + 3];
    const float* mn = (const float*)d_in[1 + i * 6 + 4];
    const float* vr = (const float*)d_in[1 + i * 6 + 5];
    int total = LCout[i] * Kpad[i];
    prep_weights<<<(total + 255) / 256, 256, 0, stream>>>(
        w, b, g, be, mn, vr, wp[i], bs[i], LCout[i], Kdim[i], Kpad[i]);
  }
  {
    int n = 20 * 3 * 256 * 256;
    cvt_f32_f16<<<(n + 255) / 256, 256, 0, stream>>>(images, imgH, n);
  }

  auto conv = [&](const _Float16* src, _Float16* dst, int li, int Nimg,
                  int H, int W, int strd, int pad, int Ctot, int Coff,
                  int grp) {
    int Cout = LCout[li], Cin = LCin[li], Kh = LKs[li];
    int OH = (H + 2 * pad - Kh) / strd + 1;
    int OW = (W + 2 * pad - Kh) / strd + 1;
    dim3 grid((unsigned)((Nimg * OH * OW + TBN - 1) / TBN),
              (unsigned)((Cout + TBM - 1) / TBM));
    conv_wmma<<<grid, 128, 0, stream>>>(src, wp[li], bs[li], dst, Nimg, Cin, H,
                                        W, Cout, Kh, Kh, strd, pad, OH, OW,
                                        Kdim[li], Kpad[li], Ctot, Coff, grp, 1);
  };

  conv(imgH, a1, 0, 20, 256, 256, 2, 3, 64, 0, 0);    // conv1 -> 128x128
  conv(a1, a2, 1, 20, 128, 128, 2, 2, 128, 0, 0);     // conv2 -> 64x64
  conv(a2, a3, 2, 20, 64, 64, 2, 2, 256, 0, 0);       // conv3 -> 32x32 feats
  {
    dim3 grid(16, 32);
    correlation_k<<<grid, 256, 0, stream>>>(a3, cat); // channels 0..440
  }
  conv(a3, cat, 3, 16, 32, 32, 1, 0, 473, 441, 4);    // conv_redir -> ch 441..472
  conv(cat, a1, 4, 16, 32, 32, 1, 1, 256, 0, 0);      // conv3_1
  conv(a1, a2, 5, 16, 32, 32, 2, 1, 512, 0, 0);       // conv4  -> 16x16
  conv(a2, a1, 6, 16, 16, 16, 1, 1, 512, 0, 0);       // conv4_1
  conv(a1, a2, 7, 16, 16, 16, 2, 1, 512, 0, 0);       // conv5  -> 8x8
  conv(a2, a1, 8, 16, 8, 8, 1, 1, 512, 0, 0);         // conv5_1
  conv(a1, a2, 9, 16, 8, 8, 2, 1, 1024, 0, 0);        // conv6  -> 4x4

  avgpool_k<<<(16 * 1024 + 255) / 256, 256, 0, stream>>>(a2, pooled,
                                                         16 * 1024, 16);
  fc_k<<<(16 * 512 + 255) / 256, 256, 0, stream>>>(
      pooled, (const float*)d_in[61], (const float*)d_in[62], fc1o, 16, 512,
      1024, 1);
  fc_k<<<(16 * 256 + 255) / 256, 256, 0, stream>>>(
      fc1o, (const float*)d_in[63], (const float*)d_in[64], (float*)d_out, 16,
      256, 512, 0);
}
